// KANMLP_23270132809814
// MI455X (gfx1250) — matrix operand order repaired and checked by reading, not compiled, not run on hardware
//
#include <hip/hip_runtime.h>

typedef _Float16 v16h __attribute__((ext_vector_type(16)));
typedef _Float16 v8h  __attribute__((ext_vector_type(8)));
typedef float    v8f  __attribute__((ext_vector_type(8)));

#define N_ROWS 8192
#define D_IN   1024
#define H_HID  4096
#define DEGP1  5
#define DEGK   4      // T1..T4 go through WMMA; T0 folded into per-channel bias
#define LN_EPS 1e-5f

// Branchless tanh: 1 - 2/(e^{2x}+1). Saturates correctly at +-1, no divergence.
__device__ __forceinline__ float fast_tanh(float x) {
    float e = __expf(2.0f * x);
    return 1.0f - 2.0f * __builtin_amdgcn_rcpf(e + 1.0f);
}

// ---------------------------------------------------------------------------
// Repack coef [IN][OUT][5] (f32) -> wT [OUT][ (k-1)*IN + i ] (f16), k = 1..4.
// ---------------------------------------------------------------------------
__global__ __launch_bounds__(256) void pack_w_kernel(const float* __restrict__ coef,
                                                     _Float16* __restrict__ wT,
                                                     int IN, int OUT) {
    size_t idx = (size_t)blockIdx.x * 256 + threadIdx.x;
    size_t total = (size_t)OUT * DEGK * IN;
    if (idx >= total) return;
    int K4  = IN * DEGK;
    int o   = (int)(idx / K4);
    int rem = (int)(idx % K4);
    int k   = rem / IN + 1;          // 1..4
    int i   = rem % IN;
    wT[idx] = (_Float16)coef[((size_t)i * OUT + o) * DEGP1 + k];
}

// bias[o] = sum_i coef[i][o][0]   (T0 == 1 contribution)
__global__ __launch_bounds__(256) void bias_kernel(const float* __restrict__ coef,
                                                   float* __restrict__ bias,
                                                   int IN, int OUT) {
    int o = blockIdx.x * 256 + threadIdx.x;
    if (o >= OUT) return;
    float s = 0.0f;
    for (int i = 0; i < IN; ++i)
        s += coef[((size_t)i * OUT + o) * DEGP1];
    bias[o] = s;
}

// --- raw 8-element loaders / unpackers (float source vs f16 source) --------
__device__ __forceinline__ void load_raw8(const float* p, float4& r0, float4& r1) {
    const float4* q = reinterpret_cast<const float4*>(p);
    r0 = q[0]; r1 = q[1];
}
__device__ __forceinline__ void load_raw8(const _Float16* p, float4& r0, float4& r1) {
    r0 = *reinterpret_cast<const float4*>(p);
    r1 = r0;
}
__device__ __forceinline__ void unpack8(const float4& r0, const float4& r1,
                                        const float*, float* o) {
    o[0] = r0.x; o[1] = r0.y; o[2] = r0.z; o[3] = r0.w;
    o[4] = r1.x; o[5] = r1.y; o[6] = r1.z; o[7] = r1.w;
}
__device__ __forceinline__ void unpack8(const float4& r0, const float4&,
                                        const _Float16*, float* o) {
    v8h h;
    __builtin_memcpy(&h, &r0, sizeof(h));
    #pragma unroll
    for (int i = 0; i < 8; ++i) o[i] = (float)h[i];
}

// ---------------------------------------------------------------------------
// Fused Chebyshev-KAN GEMM.
//   out[N][Nout] = cheb_{1..4}(act(src))[N][4*Cin] * wT^T + bias
// Workgroup tile 128(M) x 128(N); 8 waves, each a 2(M)x4(N) block of tiles.
// B tiles: GLOBAL_LOAD_ASYNC_TO_LDS_B128 (ASYNCcnt), no VGPR staging.
// A tiles: global->VGPR->tanh/cheb->LDS (b128 stores), double buffered.
// ---------------------------------------------------------------------------
template <bool APPLY_TANH, typename SrcT>
__global__ __launch_bounds__(256, 1) void kan_gemm_kernel(const SrcT* __restrict__ src,
                                                          const _Float16* __restrict__ wT,
                                                          const float* __restrict__ bias,
                                                          float* __restrict__ out,
                                                          int Cin, int Nout) {
    __shared__ alignas(16) _Float16 Alds[2][DEGK * 128 * 32]; // 2 x 32 KB
    __shared__ alignas(16) _Float16 Blds[2][128 * DEGK * 32]; // 2 x 32 KB

    const int tid  = threadIdx.x;
    const int lane = tid & 31;
    const int wave = tid >> 5;
    const int mW   = (wave >> 1) * 2;   // first of two M tiles (0,2,4,6)
    const int nW   = (wave & 1) * 4;    // first of four N tiles (0,4)
    const int lr   = lane & 15;
    const int hi   = lane >> 4;

    const int rowBase = blockIdx.y * 128;
    const int colBase = blockIdx.x * 128;
    const int K4 = Cin * DEGK;

    v8f acc[2][4] = {};

    float4 aRaw[2][2];   // 2 tasks x 8 source values

    // B: 2048 16B chunks (128 cols x 4 planes x 4 segs), 8 asyncs/thread.
    auto asyncLoadB = [&](int buf, int c0) {
        #pragma unroll
        for (int e = 0; e < 8; ++e) {
            int idx = tid + e * 256;
            int o = idx >> 4, rem = idx & 15, k = rem >> 2, seg = rem & 3;
            const _Float16* g =
                &wT[(size_t)(colBase + o) * K4 + (size_t)k * Cin + c0 + seg * 8];
            unsigned l = (unsigned)(uintptr_t)&Blds[buf][o * 128 + k * 32 + seg * 8];
            asm volatile("global_load_async_to_lds_b128 %0, %1, off"
                         :: "v"(l), "v"(g) : "memory");
        }
    };
    // A: 512 tasks (128 rows x 4 groups of 8 ch), 2/thread.
    auto loadRegsA = [&](int c0) {
        #pragma unroll
        for (int e = 0; e < 2; ++e) {
            int idx = tid * 2 + e;
            int r = idx >> 2, g = idx & 3;
            load_raw8(&src[(size_t)(rowBase + r) * Cin + c0 + g * 8],
                      aRaw[e][0], aRaw[e][1]);
        }
    };
    auto storeA = [&](int buf) {
        #pragma unroll
        for (int e = 0; e < 2; ++e) {
            int idx = tid * 2 + e;
            int r = idx >> 2, g = idx & 3;
            float t[8];
            unpack8(aRaw[e][0], aRaw[e][1], (const SrcT*)nullptr, t);
            v8h p1, p2, p3, p4;
            #pragma unroll
            for (int i = 0; i < 8; ++i) {
                float tv = APPLY_TANH ? fast_tanh(t[i]) : t[i];
                float c2 = 2.0f * tv * tv - 1.0f;
                float c3 = 2.0f * tv * c2 - tv;
                float c4 = 2.0f * tv * c3 - c2;
                p1[i] = (_Float16)tv; p2[i] = (_Float16)c2;
                p3[i] = (_Float16)c3; p4[i] = (_Float16)c4;
            }
            _Float16* base = &Alds[buf][r * 32 + g * 8];
            *reinterpret_cast<v8h*>(base + 0 * 4096) = p1;
            *reinterpret_cast<v8h*>(base + 1 * 4096) = p2;
            *reinterpret_cast<v8h*>(base + 2 * 4096) = p3;
            *reinterpret_cast<v8h*>(base + 3 * 4096) = p4;
        }
    };
    auto compute = [&](int buf) {
        #pragma unroll
        for (int ks = 0; ks < DEGK; ++ks) {
            v16h bf[4], af[2];
            #pragma unroll
            for (int ni = 0; ni < 4; ++ni) {
                const _Float16* bp =
                    &Blds[buf][((nW + ni) * 16 + lr) * 128 + ks * 32 + hi * 16];
                reinterpret_cast<v8h*>(&bf[ni])[0] = reinterpret_cast<const v8h*>(bp)[0];
                reinterpret_cast<v8h*>(&bf[ni])[1] = reinterpret_cast<const v8h*>(bp)[1];
            }
            #pragma unroll
            for (int mi = 0; mi < 2; ++mi) {
                const _Float16* ap =
                    &Alds[buf][ks * 4096 + ((mW + mi) * 16 + lr) * 32 + hi * 16];
                reinterpret_cast<v8h*>(&af[mi])[0] = *reinterpret_cast<const v8h*>(ap);
                reinterpret_cast<v8h*>(&af[mi])[1] = *reinterpret_cast<const v8h*>(ap + 8);
            }
            #pragma unroll
            for (int mi = 0; mi < 2; ++mi)
                #pragma unroll
                for (int ni = 0; ni < 4; ++ni)
                    acc[mi][ni] = __builtin_amdgcn_wmma_f32_16x16x32_f16(
                        false, af[mi], false, bf[ni], (short)0, acc[mi][ni], false, false);
        }
    };

    // ---- pipeline: async-B(n+1) + loadA(n+1) -> wmma(n) -> storeA(n+1) ----
    asyncLoadB(0, 0);
    loadRegsA(0);
    storeA(0);
    asm volatile("s_wait_asynccnt 0x0" ::: "memory");
    __syncthreads();

    const int nCh = Cin >> 5;
    for (int ch = 0; ch < nCh; ++ch) {
        const int buf = ch & 1;
        if (ch + 1 < nCh) {
            asyncLoadB(buf ^ 1, (ch + 1) << 5);
            loadRegsA((ch + 1) << 5);
        }
        compute(buf);
        if (ch + 1 < nCh) storeA(buf ^ 1);
        asm volatile("s_wait_asynccnt 0x0" ::: "memory");
        __syncthreads();
    }

    // ---- epilogue: add T0 bias; C layout: VGPR v -> M = v + 8*hi, lane -> N
    #pragma unroll
    for (int ni = 0; ni < 4; ++ni) {
        const int col = colBase + (nW + ni) * 16 + lr;
        const float bv = bias[col];
        #pragma unroll
        for (int mi = 0; mi < 2; ++mi) {
            #pragma unroll
            for (int v = 0; v < 8; ++v) {
                int r = rowBase + (mW + mi) * 16 + v + hi * 8;
                out[(size_t)r * Nout + col] = acc[mi][ni][v] + bv;
            }
        }
    }
}

// ---------------------------------------------------------------------------
// Row-wise LayerNorm + tanh; writes f16 for GEMM2's A-source.
// ---------------------------------------------------------------------------
__global__ __launch_bounds__(256) void ln_tanh_kernel(const float* __restrict__ h,
                                                      const float* __restrict__ gamma,
                                                      const float* __restrict__ beta,
                                                      _Float16* __restrict__ hn) {
    __shared__ float red[256];
    const int row = blockIdx.x;
    const int tid = threadIdx.x;
    const float* hr = h + (size_t)row * H_HID;

    float s = 0.f, s2 = 0.f;
    for (int j = tid; j < H_HID; j += 256) {
        float v = hr[j];
        s  += v;
        s2 += v * v;
    }
    red[tid] = s; __syncthreads();
    for (int off = 128; off > 0; off >>= 1) {
        if (tid < off) red[tid] += red[tid + off];
        __syncthreads();
    }
    const float mu = red[0] * (1.0f / H_HID);
    __syncthreads();
    red[tid] = s2; __syncthreads();
    for (int off = 128; off > 0; off >>= 1) {
        if (tid < off) red[tid] += red[tid + off];
        __syncthreads();
    }
    const float var  = red[0] * (1.0f / H_HID) - mu * mu;
    const float rstd = rsqrtf(var + LN_EPS);

    _Float16* hno = hn + (size_t)row * H_HID;
    for (int j = tid; j < H_HID; j += 256) {
        float v = (hr[j] - mu) * rstd * gamma[j] + beta[j];
        hno[j] = (_Float16)fast_tanh(v);
    }
}

// ---------------------------------------------------------------------------
// Launch
// ---------------------------------------------------------------------------
extern "C" void kernel_launch(void* const* d_in, const int* in_sizes, int n_in,
                              void* d_out, int out_size, void* d_ws, size_t ws_size,
                              hipStream_t stream) {
    const float* x         = (const float*)d_in[0]; // [4,2048,1024]
    const float* coef_fc   = (const float*)d_in[1]; // [1024,4096,5]
    const float* coef_proj = (const float*)d_in[2]; // [4096,1024,5]
    const float* gamma     = (const float*)d_in[3]; // [4096]
    const float* beta      = (const float*)d_in[4]; // [4096]
    float* out = (float*)d_out;                     // [4,2048,1024] f32

    // workspace partition
    char* ws = (char*)d_ws;
    const size_t W1_BYTES = (size_t)H_HID * D_IN * DEGK * sizeof(_Float16); // 32 MB
    const size_t W2_BYTES = (size_t)D_IN * H_HID * DEGK * sizeof(_Float16); // 32 MB
    const size_t B1_BYTES = (size_t)H_HID * sizeof(float);
    const size_t B2_BYTES = (size_t)D_IN * sizeof(float);
    const size_t HB_BYTES = (size_t)N_ROWS * H_HID * sizeof(float);         // 128 MB

    _Float16* W1T  = (_Float16*)(ws);
    _Float16* W2T  = (_Float16*)(ws + W1_BYTES);
    float*    Bi1  = (float*)   (ws + W1_BYTES + W2_BYTES);
    float*    Bi2  = (float*)   (ws + W1_BYTES + W2_BYTES + B1_BYTES);
    float*    Hbuf = (float*)   (ws + W1_BYTES + W2_BYTES + B1_BYTES + B2_BYTES);
    _Float16* HN   = (_Float16*)(ws + W1_BYTES + W2_BYTES + B1_BYTES + B2_BYTES + HB_BYTES);

    const size_t tot = (size_t)D_IN * H_HID * DEGK;
    const unsigned packBlocks = (unsigned)((tot + 255) / 256);
    pack_w_kernel<<<dim3(packBlocks), 256, 0, stream>>>(coef_fc,   W1T, D_IN,  H_HID);
    pack_w_kernel<<<dim3(packBlocks), 256, 0, stream>>>(coef_proj, W2T, H_HID, D_IN);
    bias_kernel<<<dim3((H_HID + 255) / 256), 256, 0, stream>>>(coef_fc,   Bi1, D_IN,  H_HID);
    bias_kernel<<<dim3((D_IN  + 255) / 256), 256, 0, stream>>>(coef_proj, Bi2, H_HID, D_IN);

    kan_gemm_kernel<true, float>
        <<<dim3(H_HID / 128, N_ROWS / 128), 256, 0, stream>>>(x, W1T, Bi1, Hbuf, D_IN, H_HID);

    ln_tanh_kernel<<<dim3(N_ROWS), 256, 0, stream>>>(Hbuf, gamma, beta, HN);

    kan_gemm_kernel<false, _Float16>
        <<<dim3(D_IN / 128, N_ROWS / 128), 256, 0, stream>>>(HN, W2T, Bi2, out, H_HID, D_IN);
}
